// InferenceLinear_27178553049572
// MI455X (gfx1250) — matrix operand order repaired and checked by reading, not compiled, not run on hardware
//
#include <hip/hip_runtime.h>

// ---------------------------------------------------------------------------
// Group-quantized linear (u8 weights, per-(out,group) scale/offset) as an
// f16-WMMA GEMM with perm-trick in-register dequantization on gfx1250.
//   M = 512, K = 4096, N = 2048, 16 groups of 256.
// Pass 1: x f32 -> f16 once into d_ws (x is reused by all 32 N-block columns).
// Pass 2: GEMM, block 256 threads (8 wave32), block tile 128M x 64N, BK=32.
//   Wave tile 32M x 32N = 4 WMMA accumulators per k-step.
//   Both operand tiles staged via global_load_async_to_lds (ASYNCcnt),
//   double-buffered; B dequant: bits(0x6400|q) == 1024+q exactly, so one
//   v_perm_b32 + one v_pk_fma_f16 per packed pair:
//     w = (1024+q)*s + (c - 1024*s) = q*s + c.
// ---------------------------------------------------------------------------

typedef __attribute__((ext_vector_type(16))) _Float16 v16h;
typedef __attribute__((ext_vector_type(8)))  _Float16 v8h;
typedef __attribute__((ext_vector_type(2)))  _Float16 v2h;
typedef __attribute__((ext_vector_type(8)))  float    v8f;
typedef __attribute__((ext_vector_type(4)))  unsigned u32x4;
typedef __attribute__((ext_vector_type(2)))  int      i32x2;
typedef __attribute__((ext_vector_type(4)))  int      i32x4;

typedef __attribute__((address_space(1))) i32x2 gl_i32x2;
typedef __attribute__((address_space(3))) i32x2 ld_i32x2;
typedef __attribute__((address_space(1))) i32x4 gl_i32x4;
typedef __attribute__((address_space(3))) i32x4 ld_i32x4;

#define IN_F    4096
#define OUT_F   2048
#define MROWS   512
#define NGROUPS 16

#define BM 128
#define BN 64
#define BK 32
#define KSTEPS (IN_F / BK)   // 128 k-steps, 8 per quant group

#if __has_builtin(__builtin_amdgcn_global_load_async_to_lds_b64)
#define USE_ASYNC_LDS 1
#endif
#if __has_builtin(__builtin_amdgcn_global_load_async_to_lds_b128)
#define USE_ASYNC_B128 1
#endif

__device__ __forceinline__ void wait_async_all() {
#ifdef USE_ASYNC_LDS
#if __has_builtin(__builtin_amdgcn_s_wait_asynccnt)
  __builtin_amdgcn_s_wait_asynccnt(0);
#else
  asm volatile("s_wait_asynccnt 0" ::: "memory");
#endif
#endif
}

// ---- pass 1: one-shot x conversion (f32 -> f16), 8 elems/thread ------------
__global__ __launch_bounds__(256)
void convert_x_f16(const float* __restrict__ x, _Float16* __restrict__ xh) {
  const size_t i = ((size_t)blockIdx.x * 256 + threadIdx.x) * 8;
  float4 f0 = *(const float4*)(x + i);
  float4 f1 = *(const float4*)(x + i + 4);
  v8h h;
  h[0] = (_Float16)f0.x; h[1] = (_Float16)f0.y;
  h[2] = (_Float16)f0.z; h[3] = (_Float16)f0.w;
  h[4] = (_Float16)f1.x; h[5] = (_Float16)f1.y;
  h[6] = (_Float16)f1.z; h[7] = (_Float16)f1.w;
  *(v8h*)(xh + i) = h;
}

// ---- pass 2: dequant GEMM --------------------------------------------------
template <bool PRECONV>
__global__ __launch_bounds__(256)
void InferenceLinear_dq_wmma(const float* __restrict__ x,
                             const _Float16* __restrict__ xh,
                             const unsigned char* __restrict__ qw,
                             const float* __restrict__ qrange,
                             const float* __restrict__ qmin,
                             float* __restrict__ out)
{
  __shared__ _Float16      xs[2][BM * BK];   // 2 * 8 KB, f16 [m][k]
  __shared__ unsigned char qs[2][BN * BK];   // 2 * 2 KB, u8  [n][k]

  const int tid  = threadIdx.x;
  const int lane = tid & 31;
  const int wave = tid >> 5;
  const int wm   = wave & 3;    // M sub-block: rows [wm*32, wm*32+32)
  const int wn   = wave >> 2;   // N sub-block: cols [wn*32, wn*32+32)

  const int n0 = blockIdx.x * BN;
  const int m0 = blockIdx.y * BM;

  // Staging: x tile 128x32 (16 elems/thread); q tile 64x32 u8 (8 B/thread)
  const int xr = tid >> 1;          // 0..127
  const int xc = (tid & 1) * 16;    // 0 or 16 (f16 elements)
  const int qn = tid >> 2;          // 0..63
  const int qc = (tid & 3) * 8;     // 0,8,16,24

  v8f acc00 = {}, acc01 = {}, acc10 = {}, acc11 = {};

  auto stage = [&](int step, int buf) {
    const int k0 = step * BK;
    _Float16* xlp = &xs[buf][xr * BK + xc];
    if (PRECONV) {
      // pure async copy of pre-converted f16 x (no VGPR round-trip)
      const _Float16* xp = xh + (size_t)(m0 + xr) * IN_F + (k0 + xc);
#if defined(USE_ASYNC_B128)
      __builtin_amdgcn_global_load_async_to_lds_b128(
          (gl_i32x4*)xp, (ld_i32x4*)xlp, 0, 0);
      __builtin_amdgcn_global_load_async_to_lds_b128(
          (gl_i32x4*)(xp + 8), (ld_i32x4*)(xlp + 8), 0, 0);
#elif defined(USE_ASYNC_LDS)
      __builtin_amdgcn_global_load_async_to_lds_b64(
          (gl_i32x2*)xp, (ld_i32x2*)xlp, 0, 0);
      __builtin_amdgcn_global_load_async_to_lds_b64(
          (gl_i32x2*)(xp + 4), (ld_i32x2*)(xlp + 4), 0, 0);
      __builtin_amdgcn_global_load_async_to_lds_b64(
          (gl_i32x2*)(xp + 8), (ld_i32x2*)(xlp + 8), 0, 0);
      __builtin_amdgcn_global_load_async_to_lds_b64(
          (gl_i32x2*)(xp + 12), (ld_i32x2*)(xlp + 12), 0, 0);
#else
      *(v8h*)(xlp)     = *(const v8h*)(xp);
      *(v8h*)(xlp + 8) = *(const v8h*)(xp + 8);
#endif
    } else {
      // fallback: in-kernel f32 -> f16 conversion
      const float* xp = x + (size_t)(m0 + xr) * IN_F + (k0 + xc);
      float4 f0 = *(const float4*)(xp);
      float4 f1 = *(const float4*)(xp + 4);
      float4 f2 = *(const float4*)(xp + 8);
      float4 f3 = *(const float4*)(xp + 12);
      v8h h0, h1;
      h0[0] = (_Float16)f0.x; h0[1] = (_Float16)f0.y;
      h0[2] = (_Float16)f0.z; h0[3] = (_Float16)f0.w;
      h0[4] = (_Float16)f1.x; h0[5] = (_Float16)f1.y;
      h0[6] = (_Float16)f1.z; h0[7] = (_Float16)f1.w;
      h1[0] = (_Float16)f2.x; h1[1] = (_Float16)f2.y;
      h1[2] = (_Float16)f2.z; h1[3] = (_Float16)f2.w;
      h1[4] = (_Float16)f3.x; h1[5] = (_Float16)f3.y;
      h1[6] = (_Float16)f3.z; h1[7] = (_Float16)f3.w;
      *(v8h*)(xlp)     = h0;
      *(v8h*)(xlp + 8) = h1;
    }
    // ---- weights: raw u8, async copy global -> LDS (ASYNCcnt path)
    const unsigned char* qp = qw + (size_t)(n0 + qn) * IN_F + (k0 + qc);
    unsigned char* lp = &qs[buf][qn * BK + qc];
#ifdef USE_ASYNC_LDS
    __builtin_amdgcn_global_load_async_to_lds_b64(
        (gl_i32x2*)qp, (ld_i32x2*)lp, 0, 0);
#else
    *(unsigned long long*)lp = *(const unsigned long long*)qp;
#endif
  };

  // perm-trick dequant of one B fragment (16 contiguous u8 from LDS).
  auto dq = [&](const unsigned char* bbase, v2h sv, v2h cv) -> v16h {
    u32x4 qv = *(const u32x4*)bbase;                 // ds_load_b128
    union { unsigned u[8]; v16h v; } r;
#pragma unroll
    for (int d = 0; d < 4; ++d) {
      // out bytes [q(2k), 0x64, q(2k+1), 0x64] == f16 pair (1024+q, 1024+q')
      unsigned plo = __builtin_amdgcn_perm(0x64646464u, qv[d], 0x04010400u);
      unsigned phi = __builtin_amdgcn_perm(0x64646464u, qv[d], 0x04030402u);
      v2h a = __builtin_bit_cast(v2h, plo) * sv + cv;   // v_pk_fma_f16
      v2h b = __builtin_bit_cast(v2h, phi) * sv + cv;
      r.u[2 * d]     = __builtin_bit_cast(unsigned, a);
      r.u[2 * d + 1] = __builtin_bit_cast(unsigned, b);
    }
    return r.v;
  };

  // A fragment: lanes<16 hold K 0..15, lanes>=16 hold K 16..31
  auto ldA = [&](const _Float16* xbase, int a) -> v16h {
    const int arow = (wm * 32 + a * 16 + (lane & 15)) * BK + ((lane & 16) ? 16 : 0);
    v8h lo = *(const v8h*)(xbase + arow);
    v8h hi = *(const v8h*)(xbase + arow + 8);
    return __builtin_shufflevector(lo, hi,
        0, 1, 2, 3, 4, 5, 6, 7, 8, 9, 10, 11, 12, 13, 14, 15);
  };

  stage(0, 0);

  const int nlo = lane & 15;
  const int o0  = n0 + wn * 32 + nlo;    // B tile 0 column
  const int o1  = o0 + 16;               // B tile 1 column

  for (int g = 0; g < NGROUPS; ++g) {
    // Per-(column, group) dequant constants; absorb the 1024 magic bias.
    const float s0f = qrange[o0 * NGROUPS + g];
    const float c0f = qmin  [o0 * NGROUPS + g];
    const float s1f = qrange[o1 * NGROUPS + g];
    const float c1f = qmin  [o1 * NGROUPS + g];
    const _Float16 s0h = (_Float16)s0f, c0h = (_Float16)(c0f - 1024.0f * s0f);
    const _Float16 s1h = (_Float16)s1f, c1h = (_Float16)(c1f - 1024.0f * s1f);
    const v2h sv0 = {s0h, s0h}, cv0 = {c0h, c0h};
    const v2h sv1 = {s1h, s1h}, cv1 = {c1h, c1h};

#pragma unroll 2
    for (int kk = 0; kk < 8; ++kk) {
      const int step = g * 8 + kk;
      const int buf  = step & 1;

      wait_async_all();         // own async copies done before signaling
      __syncthreads();          // tile[buf] ready; tile[buf^1] free to refill

      if (step + 1 < KSTEPS) stage(step + 1, buf ^ 1);

      const _Float16* xbase = xs[buf];
      v16h a0 = ldA(xbase, 0);
      v16h a1 = ldA(xbase, 1);

      const int khalf = (lane & 16) ? 16 : 0;
      {
        const unsigned char* bb = &qs[buf][(wn * 32 + nlo) * BK + khalf];
        v16h b0 = dq(bb, sv0, cv0);
        acc00 = __builtin_amdgcn_wmma_f32_16x16x32_f16(
            false, a0, false, b0, (short)0, acc00, false, false);
        acc10 = __builtin_amdgcn_wmma_f32_16x16x32_f16(
            false, a1, false, b0, (short)0, acc10, false, false);
      }
      {
        const unsigned char* bb = &qs[buf][(wn * 32 + 16 + nlo) * BK + khalf];
        v16h b1 = dq(bb, sv1, cv1);
        acc01 = __builtin_amdgcn_wmma_f32_16x16x32_f16(
            false, a0, false, b1, (short)0, acc01, false, false);
        acc11 = __builtin_amdgcn_wmma_f32_16x16x32_f16(
            false, a1, false, b1, (short)0, acc11, false, false);
      }
    }
  }

  // Epilogue: C/D layout — lane holds column n, VGPR r is row M=r (+8 upper half)
  const int mb0 = m0 + wm * 32 + ((lane & 16) ? 8 : 0);
#pragma unroll
  for (int r = 0; r < 8; ++r) {
    out[(size_t)(mb0 + r) * OUT_F + o0]      = acc00[r];
    out[(size_t)(mb0 + r) * OUT_F + o1]      = acc01[r];
    out[(size_t)(mb0 + 16 + r) * OUT_F + o0] = acc10[r];
    out[(size_t)(mb0 + 16 + r) * OUT_F + o1] = acc11[r];
  }
}

extern "C" void kernel_launch(void* const* d_in, const int* in_sizes, int n_in,
                              void* d_out, int out_size, void* d_ws, size_t ws_size,
                              hipStream_t stream) {
  const float*         x      = (const float*)d_in[0];
  const unsigned char* qw     = (const unsigned char*)d_in[1];
  const float*         qrange = (const float*)d_in[2];
  const float*         qmin   = (const float*)d_in[3];
  float*               out    = (float*)d_out;

  dim3 grid(OUT_F / BN, MROWS / BM);   // 32 x 4 = 128 blocks
  const size_t xh_bytes = (size_t)MROWS * IN_F * sizeof(_Float16);   // 4 MB

  if (ws_size >= xh_bytes) {
    _Float16* xh = (_Float16*)d_ws;
    convert_x_f16<<<(MROWS * IN_F) / (8 * 256), 256, 0, stream>>>(x, xh);
    InferenceLinear_dq_wmma<true><<<grid, 256, 0, stream>>>(
        x, xh, qw, qrange, qmin, out);
  } else {
    InferenceLinear_dq_wmma<false><<<grid, 256, 0, stream>>>(
        x, nullptr, qw, qrange, qmin, out);
  }
}
